// AdditiveAttnLayer_36266703847883
// MI455X (gfx1250) — matrix operand order repaired and checked by reading, not compiled, not run on hardware
//
#include <hip/hip_runtime.h>
#include <hip/hip_bf16.h>

// ---------------------------------------------------------------------------
// GRIT-style additive attention layer for MI455X (gfx1250, wave32, WMMA).
// All heavy GEMMs use v_wmma_f32_16x16x32_f16 (f16 inputs, f32 accumulate).
// ---------------------------------------------------------------------------

typedef __attribute__((ext_vector_type(16))) _Float16     v16h;
typedef __attribute__((ext_vector_type(8)))  float        v8f;
typedef __attribute__((ext_vector_type(8)))  unsigned int v8u;

union Frag { v8u u; v16h h; };

#define H_HEADS 8
#define D_DIM   32
#define IN_DIM  256

// ---------------- conversion / packing kernels ----------------

__global__ void k_to_half(const float* __restrict__ x, _Float16* __restrict__ y, size_t n) {
    size_t i = (size_t)blockIdx.x * blockDim.x + threadIdx.x;
    if (i < n) y[i] = (_Float16)x[i];
}

// W stored [in][out] row-major -> Wt stored [out][in] f16 (so B fragment rows
// are contiguous along K, same access pattern as A rows).
__global__ void k_pack_wt(const float* __restrict__ w, _Float16* __restrict__ wt,
                          int in_dim, int out_dim) {
    int idx = blockIdx.x * blockDim.x + threadIdx.x;
    if (idx >= in_dim * out_dim) return;
    int o = idx / in_dim;
    int i = idx - o * in_dim;
    wt[idx] = (_Float16)w[(size_t)i * out_dim + o];
}

// ---------------- WMMA GEMM: C[M,NCOLS] = A[M,K] @ Bt^T + bias (+epilogue) --
// MODE 0: f32 out = C+bias
// MODE 1: f16 out = C+bias
// MODE 2: f16 out = relu(C+bias)
// MODE 3: f32 out = C+bias+res
template<int K, int NCOLS, int MODE>
__global__ __launch_bounds__(256)
void k_gemm(const _Float16* __restrict__ A, const _Float16* __restrict__ Bt,
            const float* __restrict__ bias, const float* __restrict__ res,
            float* __restrict__ outF, _Float16* __restrict__ outH, int Mtiles) {
    constexpr int NT   = 4;                 // 16-col tiles per wave (64 cols)
    constexpr int NBLK = NCOLS / (16 * NT); // column blocks
    int gw = blockIdx.x * (blockDim.x >> 5) + (threadIdx.x >> 5);
    if (gw >= Mtiles * NBLK) return;
    int mt   = gw / NBLK;
    int nb   = gw - mt * NBLK;
    int lane = threadIdx.x & 31;
    int l15  = lane & 15;
    int hi   = lane >> 4;

    // A row for this lane (16-bit A 16x32 layout: lanes0-15 M=0..15 K{0..7,16..23},
    // lanes16-31 same M range, K{8..15,24..31})
    const unsigned int* arow = (const unsigned int*)(A + (size_t)(mt * 16 + l15) * K);
    const unsigned int* brow[NT];
#pragma unroll
    for (int t = 0; t < NT; ++t)
        brow[t] = (const unsigned int*)(Bt + (size_t)(nb * NT * 16 + t * 16 + l15) * K);

    // pair (uint) offsets within a 32-wide K step, per ISA 7.12.2 layouts
    int kA[8], kB[8];
#pragma unroll
    for (int v = 0; v < 8; ++v) {
        kA[v] = (hi ? 4 : 0) + ((v < 4) ? 0 : 8) + (v & 3);
        kB[v] = (hi ? 4 : 0) + ((v < 4) ? 0 : 8) + (v & 3);
    }

    v8f zero = {0.f, 0.f, 0.f, 0.f, 0.f, 0.f, 0.f, 0.f};
    v8f acc[NT];
#pragma unroll
    for (int t = 0; t < NT; ++t) acc[t] = zero;

    for (int k0 = 0; k0 < K / 2; k0 += 16) { // 16 uints = 32 halves per step
        Frag a;
#pragma unroll
        for (int v = 0; v < 8; ++v) a.u[v] = arow[k0 + kA[v]];
#pragma unroll
        for (int t = 0; t < NT; ++t) {
            Frag b;
#pragma unroll
            for (int v = 0; v < 8; ++v) b.u[v] = brow[t][k0 + kB[v]];
            acc[t] = __builtin_amdgcn_wmma_f32_16x16x32_f16(
                false, a.h, false, b.h, (short)0, acc[t], false, false);
        }
    }

    // D layout: lanes0-15 N=lane, M=vgpr; lanes16-31 N=lane-16, M=vgpr+8
    int nBase = nb * NT * 16;
    int mBase = mt * 16 + (hi ? 8 : 0);
#pragma unroll
    for (int t = 0; t < NT; ++t) {
        int n = nBase + t * 16 + l15;
        float bv = bias[n];
#pragma unroll
        for (int v = 0; v < 8; ++v) {
            size_t oi = (size_t)(mBase + v) * NCOLS + n;
            float val = acc[t][v] + bv;
            if (MODE == 0)      outF[oi] = val;
            else if (MODE == 1) outH[oi] = (_Float16)val;
            else if (MODE == 2) outH[oi] = (_Float16)fmaxf(val, 0.f);
            else                outF[oi] = val + res[oi];
        }
    }
}

// ---------------- edge kernels -------------------------------------------
// Block = one edge (256 threads = 8 waves); wave = head; lane = head-dim.

__global__ __launch_bounds__(256)
void k_edge_score(const _Float16* __restrict__ Exh, const float* __restrict__ Kh,
                  const float* __restrict__ Qh, const int* __restrict__ src,
                  const int* __restrict__ dst, const float* __restrict__ Aw,
                  _Float16* __restrict__ connh, float* __restrict__ expScore,
                  float* __restrict__ expSum) {
    int e = blockIdx.x;
    int j = threadIdx.x;        // 0..255
    int h = j >> 5;
    int d = j & 31;
    int s = src[e], t = dst[e];

    float ex1 = (float)Exh[(size_t)e * 512 + h * 64 + d];
    float ex2 = (float)Exh[(size_t)e * 512 + h * 64 + 32 + d];
    float s2  = ex1 * ex2;
    float sc2 = copysignf(sqrtf(fabsf(s2)), s2);  // sign(s2)*sqrt(|s2|)
    float conn = Kh[(size_t)s * 256 + j] + Qh[(size_t)t * 256 + j] + sc2;
    conn = fmaxf(conn, 0.f);
    connh[(size_t)e * 256 + j] = (_Float16)conn;

    float p = conn * Aw[d * H_HEADS + h];
#pragma unroll
    for (int off = 16; off > 0; off >>= 1) p += __shfl_xor(p, off, 32);
    if (d == 0) {
        float score = fminf(fmaxf(p, -5.f), 5.f);  // clamp => exp safe w/o max-sub
        float es = expf(score);
        expScore[(size_t)e * H_HEADS + h] = es;
        atomicAdd(&expSum[(size_t)t * H_HEADS + h], es);
    }
}

__global__ __launch_bounds__(256)
void k_edge_scatter(const _Float16* __restrict__ connh, const float* __restrict__ Vh,
                    const int* __restrict__ src, const int* __restrict__ dst,
                    const float* __restrict__ expScore, const float* __restrict__ expSum,
                    float* __restrict__ msg, float* __restrict__ rowW) {
    int e = blockIdx.x;
    int j = threadIdx.x;
    int h = j >> 5;
    int s = src[e], t = dst[e];
    float attn = expScore[(size_t)e * H_HEADS + h] /
                 (expSum[(size_t)t * H_HEADS + h] + 1e-16f);
    float conn = (float)connh[(size_t)e * 256 + j];
    float vh   = Vh[(size_t)s * 256 + j];
    atomicAdd(&msg[(size_t)t * 256 + j], vh * attn);
    atomicAdd(&rowW[(size_t)t * 256 + j], conn * attn);
}

// Vo = Qh + msg + rowW @ VeRow (per head), then degree scaler, emit f16.
__global__ __launch_bounds__(256)
void k_combine_scale(const float* __restrict__ Qh, const float* __restrict__ msg,
                     const float* __restrict__ rowW, const float* __restrict__ VeRow,
                     const float* __restrict__ deg_coef, const float* __restrict__ log_deg,
                     _Float16* __restrict__ nhs) {
    int n = blockIdx.x;
    int j = threadIdx.x;
    int h = j >> 5;
    int c = j & 31;
    __shared__ float sm[256];
    sm[j] = rowW[(size_t)n * 256 + j];
    __syncthreads();
    float acc = Qh[(size_t)n * 256 + j] + msg[(size_t)n * 256 + j];
    float rv = 0.f;
#pragma unroll 8
    for (int d = 0; d < 32; ++d)
        rv += sm[h * 32 + d] * VeRow[(size_t)d * 256 + h * 32 + c];  // VeRow[d,h,c]
    acc += rv;
    float ld = log_deg[n];
    float val = acc * deg_coef[2 * j] + acc * ld * deg_coef[2 * j + 1];
    nhs[(size_t)n * 256 + j] = (_Float16)val;
}

// ---------------- batch-norm helpers -------------------------------------

__global__ __launch_bounds__(256)
void k_col_stats(const float* __restrict__ x, float* __restrict__ s1,
                 float* __restrict__ s2, int rows, int rowsPerBlock) {
    int col = threadIdx.x;                  // 256 cols
    int r0 = blockIdx.x * rowsPerBlock;
    int r1 = min(r0 + rowsPerBlock, rows);
    float a = 0.f, b = 0.f;
    for (int r = r0; r < r1; ++r) {
        float v = x[(size_t)r * 256 + col];
        a += v;
        b += v * v;
    }
    atomicAdd(&s1[col], a);
    atomicAdd(&s2[col], b);
}

__global__ __launch_bounds__(256)
void k_bn_apply(const float* __restrict__ x, const float* __restrict__ s1,
                const float* __restrict__ s2, const float* __restrict__ g,
                const float* __restrict__ b, float invn,
                float* __restrict__ outF, _Float16* __restrict__ outH, size_t total) {
    size_t i = (size_t)blockIdx.x * blockDim.x + threadIdx.x;
    if (i >= total) return;
    int col = (int)(i & 255);
    float mean = s1[col] * invn;
    float var  = fmaf(-mean, mean, s2[col] * invn);   // biased variance
    float sc   = g[col] * rsqrtf(var + 1e-5f);
    float v    = (x[i] - mean) * sc + b[col];
    if (outF) outF[i] = v;
    if (outH) outH[i] = (_Float16)v;
}

// ---------------- host orchestration -------------------------------------

extern "C" void kernel_launch(void* const* d_in, const int* in_sizes, int n_in,
                              void* d_out, int out_size, void* d_ws, size_t ws_size,
                              hipStream_t stream) {
    (void)n_in; (void)out_size; (void)ws_size;
    const float* x         = (const float*)d_in[0];
    const float* edge_attr = (const float*)d_in[1];
    const float* log_deg   = (const float*)d_in[2];
    const int*   ei        = (const int*)d_in[3];
    const float* Qw = (const float*)d_in[4];  const float* Qb = (const float*)d_in[5];
    const float* Kw = (const float*)d_in[6];  const float* Kb = (const float*)d_in[7];
    const float* Ew = (const float*)d_in[8];  const float* Eb = (const float*)d_in[9];
    const float* Vw = (const float*)d_in[10]; const float* Vb = (const float*)d_in[11];
    const float* Aw = (const float*)d_in[12]; const float* VeRow = (const float*)d_in[13];
    const float* deg_coef = (const float*)d_in[14];
    const float* NowW = (const float*)d_in[15]; const float* Nob = (const float*)d_in[16];
    const float* EowW = (const float*)d_in[17]; const float* Eob = (const float*)d_in[18];
    const float* bn1n_g = (const float*)d_in[19]; const float* bn1n_b = (const float*)d_in[20];
    const float* bn1e_g = (const float*)d_in[21]; const float* bn1e_b = (const float*)d_in[22];
    const float* f1w = (const float*)d_in[23]; const float* f1b = (const float*)d_in[24];
    const float* f2w = (const float*)d_in[25]; const float* f2b = (const float*)d_in[26];
    const float* bn2_g = (const float*)d_in[27]; const float* bn2_b = (const float*)d_in[28];

    const int N = in_sizes[0] / IN_DIM;
    const int E = in_sizes[1] / IN_DIM;
    const int* src = ei;
    const int* dstI = ei + E;

    // ---- workspace carve-out (aliasing: Exh<->eh_res, eah<->connh) ----
    char* base = (char*)d_ws;
    size_t off = 0;
    auto alloc = [&](size_t bytes) -> void* {
        void* p = base + off;
        off += (bytes + 255) & ~(size_t)255;
        return p;
    };
    _Float16* xh    = (_Float16*)alloc((size_t)N * 256 * 2);
    _Float16* eah   = (_Float16*)alloc((size_t)E * 256 * 2);   // later reused as connh
    void*     bigE  = alloc((size_t)E * 512 * 2);               // Exh (f16) / eh_res (f32)
    _Float16* Exh   = (_Float16*)bigE;
    float*    eh_res = (float*)bigE;
    float* QhB = (float*)alloc((size_t)N * 256 * 4);
    float* KhB = (float*)alloc((size_t)N * 256 * 4);
    float* VhB = (float*)alloc((size_t)N * 256 * 4);
    float* expScore = (float*)alloc((size_t)E * H_HEADS * 4);
    float* expSum   = (float*)alloc((size_t)N * H_HEADS * 4);
    float* msg  = (float*)alloc((size_t)N * 256 * 4);
    float* rowW = (float*)alloc((size_t)N * 256 * 4);
    _Float16* nhs  = (_Float16*)alloc((size_t)N * 256 * 2);
    float* nh_res  = (float*)alloc((size_t)N * 256 * 4);
    float* nh1F    = (float*)alloc((size_t)N * 256 * 4);
    _Float16* nh1H = (_Float16*)alloc((size_t)N * 256 * 2);
    _Float16* hbuf = (_Float16*)alloc((size_t)N * 512 * 2);
    float* ybuf    = (float*)alloc((size_t)N * 256 * 4);
    _Float16* Qt   = (_Float16*)alloc(256 * 256 * 2);
    _Float16* Kt   = (_Float16*)alloc(256 * 256 * 2);
    _Float16* Vt   = (_Float16*)alloc(256 * 256 * 2);
    _Float16* Ewt  = (_Float16*)alloc(512 * 256 * 2);
    _Float16* Nowt = (_Float16*)alloc(256 * 256 * 2);
    _Float16* Eowt = (_Float16*)alloc(256 * 256 * 2);
    _Float16* f1t  = (_Float16*)alloc(512 * 256 * 2);
    _Float16* f2t  = (_Float16*)alloc(256 * 512 * 2);
    float* stats = (float*)alloc(6 * 256 * 4);
    float *s1n = stats, *s2n = stats + 256, *s1y = stats + 512, *s2y = stats + 768,
          *s1e = stats + 1024, *s2e = stats + 1280;
    _Float16* connh = eah;   // alias: eah dead after Ex GEMM

    // ---- zero accumulators (graph-capture safe, stream ordered) ----
    hipMemsetAsync(expSum, 0, (size_t)N * H_HEADS * 4, stream);
    hipMemsetAsync(msg,    0, (size_t)N * 256 * 4, stream);
    hipMemsetAsync(rowW,   0, (size_t)N * 256 * 4, stream);
    hipMemsetAsync(stats,  0, 6 * 256 * 4, stream);

    dim3 blk(256);
    // ---- f16 conversion + weight packing ----
    k_to_half<<<dim3(((size_t)N * 256 + 255) / 256), blk, 0, stream>>>(x, xh, (size_t)N * 256);
    k_to_half<<<dim3(((size_t)E * 256 + 255) / 256), blk, 0, stream>>>(edge_attr, eah, (size_t)E * 256);
    k_pack_wt<<<dim3((256 * 256 + 255) / 256), blk, 0, stream>>>(Qw,  Qt,  256, 256);
    k_pack_wt<<<dim3((256 * 256 + 255) / 256), blk, 0, stream>>>(Kw,  Kt,  256, 256);
    k_pack_wt<<<dim3((256 * 256 + 255) / 256), blk, 0, stream>>>(Vw,  Vt,  256, 256);
    k_pack_wt<<<dim3((256 * 512 + 255) / 256), blk, 0, stream>>>(Ew,  Ewt, 256, 512);
    k_pack_wt<<<dim3((256 * 256 + 255) / 256), blk, 0, stream>>>(NowW, Nowt, 256, 256);
    k_pack_wt<<<dim3((256 * 256 + 255) / 256), blk, 0, stream>>>(EowW, Eowt, 256, 256);
    k_pack_wt<<<dim3((256 * 512 + 255) / 256), blk, 0, stream>>>(f1w, f1t, 256, 512);
    k_pack_wt<<<dim3((512 * 256 + 255) / 256), blk, 0, stream>>>(f2w, f2t, 512, 256);

    const int MtN = N / 16, MtE = E / 16;
    auto ggrid = [](int mtiles, int nblk) { return dim3((mtiles * nblk + 7) / 8); };

    // ---- node projections Q/K/V (f32 out) ----
    k_gemm<256, 256, 0><<<ggrid(MtN, 4), blk, 0, stream>>>(xh, Qt, Qb, nullptr, QhB, nullptr, MtN);
    k_gemm<256, 256, 0><<<ggrid(MtN, 4), blk, 0, stream>>>(xh, Kt, Kb, nullptr, KhB, nullptr, MtN);
    k_gemm<256, 256, 0><<<ggrid(MtN, 4), blk, 0, stream>>>(xh, Vt, Vb, nullptr, VhB, nullptr, MtN);
    // ---- edge projection Ex (f16 out) ----
    k_gemm<256, 512, 1><<<ggrid(MtE, 8), blk, 0, stream>>>(eah, Ewt, Eb, nullptr, nullptr, Exh, MtE);

    // ---- attention: scores + segment softmax (clamp => no max pass) ----
    k_edge_score<<<dim3(E), blk, 0, stream>>>(Exh, KhB, QhB, src, dstI, Aw,
                                              connh, expScore, expSum);
    k_edge_scatter<<<dim3(E), blk, 0, stream>>>(connh, VhB, src, dstI,
                                                expScore, expSum, msg, rowW);
    k_combine_scale<<<dim3(N), blk, 0, stream>>>(QhB, msg, rowW, VeRow, deg_coef, log_deg, nhs);

    // ---- output projections with fused residual ----
    k_gemm<256, 256, 3><<<ggrid(MtN, 4), blk, 0, stream>>>(nhs, Nowt, Nob, x, nh_res, nullptr, MtN);
    k_gemm<256, 256, 3><<<ggrid(MtE, 4), blk, 0, stream>>>(connh, Eowt, Eob, edge_attr, eh_res, nullptr, MtE);

    // ---- bn1n + FFN + bn2 (node path) ----
    k_col_stats<<<dim3((N + 4095) / 4096), blk, 0, stream>>>(nh_res, s1n, s2n, N, 4096);
    k_bn_apply<<<dim3(((size_t)N * 256 + 255) / 256), blk, 0, stream>>>(
        nh_res, s1n, s2n, bn1n_g, bn1n_b, 1.f / (float)N, nh1F, nh1H, (size_t)N * 256);
    k_gemm<256, 512, 2><<<ggrid(MtN, 8), blk, 0, stream>>>(nh1H, f1t, f1b, nullptr, nullptr, hbuf, MtN);
    k_gemm<512, 256, 3><<<ggrid(MtN, 4), blk, 0, stream>>>(hbuf, f2t, f2b, nh1F, ybuf, nullptr, MtN);
    k_col_stats<<<dim3((N + 4095) / 4096), blk, 0, stream>>>(ybuf, s1y, s2y, N, 4096);
    k_bn_apply<<<dim3(((size_t)N * 256 + 255) / 256), blk, 0, stream>>>(
        ybuf, s1y, s2y, bn2_g, bn2_b, 1.f / (float)N,
        (float*)d_out, nullptr, (size_t)N * 256);

    // ---- bn1e (edge path) ----
    k_col_stats<<<dim3((E + 4095) / 4096), blk, 0, stream>>>(eh_res, s1e, s2e, E, 4096);
    k_bn_apply<<<dim3(((size_t)E * 256 + 255) / 256), blk, 0, stream>>>(
        eh_res, s1e, s2e, bn1e_g, bn1e_b, 1.f / (float)E,
        (float*)d_out + (size_t)N * 256, nullptr, (size_t)E * 256);
}
